// Rwkv5SelfAttention_59330678227661
// MI455X (gfx1250) — matrix operand order repaired and checked by reading, not compiled, not run on hardware
//
#include <hip/hip_runtime.h>

// ---------------------------------------------------------------------------
// RWKV5 attention for MI455X (gfx1250, wave32, WMMA + async-LDS + TDM).
// ---------------------------------------------------------------------------

typedef __attribute__((ext_vector_type(16))) __bf16 v16bf;
typedef __attribute__((ext_vector_type(8)))  float  v8f;
typedef __attribute__((ext_vector_type(4)))  unsigned int u32x4;
typedef __attribute__((ext_vector_type(4)))  int    i32x4;
typedef __attribute__((ext_vector_type(8)))  int    i32x8;

#define HIDN 4096
#define SEQ  1024
#define NH   64
#define HD   64
#define CT   32               // scan chunk length (matches WMMA K for bf16)
#define NCHUNK (SEQ / CT)

// ---------------------------------------------------------------------------
// WMMA helper: D = A(16x32 bf16) x B(32x16 bf16) + C(16x16 f32)
// ---------------------------------------------------------------------------
static __device__ __forceinline__ v8f wmma_bf16(v16bf a, v16bf b, v8f c) {
  return __builtin_amdgcn_wmma_f32_16x16x32_bf16(false, a, false, b,
                                                 (short)0, c, false, false);
}

// A-fragment (16xK tile, K=32) from a row-major [16+][stride] bf16 array.
// Layout: lane<16: M=lane, elems 0..15 = K0..15
//         lane>=16: M=lane-16, elems 0..15 = K16..31
static __device__ __forceinline__ v16bf frag_a(const __bf16* base, int stride,
                                               int lane) {
  const __bf16* p = base + (lane & 15) * stride + ((lane >> 4) & 1) * 16;
  union { int4 q[2]; v16bf v; } u;
  u.q[0] = *(const int4*)(p);
  u.q[1] = *(const int4*)(p + 8);
  return u.v;
}

// B-fragment (KxN tile, K=32, N=16) from an array stored transposed [N][K].
// lane<16: N=lane, elems = K0..15 contiguous; lane>=16: N=lane-16, K16..31.
static __device__ __forceinline__ v16bf frag_b(const __bf16* baseT, int stride,
                                               int lane) {
  const __bf16* p = baseT + (lane & 15) * stride + ((lane >> 4) & 1) * 16;
  union { int4 q[2]; v16bf v; } u;
  u.q[0] = *(const int4*)(p);
  u.q[1] = *(const int4*)(p + 8);
  return u.v;
}

// Async global->LDS copy (bypasses VGPRs, tracked by ASYNCcnt).
// Generic LDS pointers carry the LDS byte offset in their low 32 bits.
static __device__ __forceinline__ void async_copy_b128(const void* g, void* l) {
  unsigned int loff = (unsigned int)(uintptr_t)l;
  asm volatile("global_load_async_to_lds_b128 %0, %1, off"
               :: "v"(loff), "v"(g) : "memory");
}
static __device__ __forceinline__ void wait_async0() {
  asm volatile("s_wait_asynccnt 0" ::: "memory");
}
static __device__ __forceinline__ void wait_async1() {  // allow 1 in flight
  asm volatile("s_wait_asynccnt 1" ::: "memory");
}

// Exchange between lane l and lane l^16 (wave32 half swap) via ds_swizzle.
static __device__ __forceinline__ float swap16(float x) {
  return __int_as_float(__builtin_amdgcn_ds_swizzle(__float_as_int(x), 0x401F));
}

// Build a B-fragment (rows = state d-index as K, cols = e) out of two WMMA
// C-layout accumulator tiles (lo = rows 0..15, hi = rows 16..31).
static __device__ __forceinline__ v16bf state_bfrag(v8f lo, v8f hi, int lane) {
  v16bf r;
  const bool upper = (lane >> 4) & 1;
#pragma unroll
  for (int i = 0; i < 8; ++i) {
    float a  = lo[i];
    float b  = hi[i];
    float as = swap16(a);
    float bs = swap16(b);
    r[i]     = (__bf16)(upper ? bs : a);   // K rows 0..7  | 16..23
    r[i + 8] = (__bf16)(upper ? b  : as);  // K rows 8..15 | 24..31
  }
  return r;
}

// ---------------------------------------------------------------------------
// Tensor Data Mover: 2D tile (rows x cols of f32, row stride in elements)
// from global memory into LDS (rows stored back-to-back). D# built per
// CDNA5 ISA 8.3/8.4: data_size=2 (4B), type=2, count=1, no padding/iterate.
// ---------------------------------------------------------------------------
#if __has_builtin(__builtin_amdgcn_tensor_load_to_lds)
#define HAVE_TDM 1
static __device__ __forceinline__ void tdm_load_2d_f32(const float* gsrc,
                                                       void* ldst, int rows,
                                                       int cols,
                                                       int row_stride) {
  unsigned long long ga = (unsigned long long)(uintptr_t)gsrc;
  unsigned int la = (unsigned int)(uintptr_t)ldst;
  u32x4 g0 = (u32x4)0u;
  g0.x = 1u;                                   // count=1, user descriptor
  g0.y = la;                                   // lds_addr
  g0.z = (unsigned int)ga;                     // global_addr lo
  g0.w = (unsigned int)(ga >> 32) | (2u << 30);// global_addr hi | type=2
  i32x8 g1 = (i32x8)0;
  g1[0] = (2 << 16);                           // data_size = 4 bytes
  g1[1] = (cols & 0xffff) << 16;               // tensor_dim0[15:0]
  g1[2] = ((rows & 0xffff) << 16) | ((cols >> 16) & 0xffff); // dim1 lo|dim0 hi
  g1[3] = ((cols & 0xffff) << 16) | ((rows >> 16) & 0xffff); // tile0 |dim1 hi
  g1[4] = (rows & 0xffff);                     // tile_dim1 (tile_dim2 = 0)
  g1[5] = row_stride;                          // tensor_dim0_stride lo32
  g1[6] = 0;
  g1[7] = 0;
  i32x4 g2 = (i32x4)0;
  i32x4 g3 = (i32x4)0;
#if defined(__clang_major__) && (__clang_major__ >= 23)
  i32x8 g4 = (i32x8)0;
  __builtin_amdgcn_tensor_load_to_lds(g0, g1, g2, g3, g4, 0);
#else
  __builtin_amdgcn_tensor_load_to_lds(g0, g1, g2, g3, 0);
#endif
}
#else
#define HAVE_TDM 0
#endif

// ---------------------------------------------------------------------------
// K0: cast f32 weight -> bf16 (weights then live in L2: 5 x 32MB < 192MB)
// ---------------------------------------------------------------------------
__global__ void __launch_bounds__(256) cast_bf16_kernel(
    const float* __restrict__ src, __bf16* __restrict__ dst) {
  int i = (blockIdx.x * 256 + threadIdx.x) * 4;
  float4 v = *(const float4*)(src + i);
  dst[i + 0] = (__bf16)v.x;
  dst[i + 1] = (__bf16)v.y;
  dst[i + 2] = (__bf16)v.z;
  dst[i + 3] = (__bf16)v.w;
}

// ---------------------------------------------------------------------------
// K1: token-shift mixing for k/v/r/g, cast to bf16 GEMM inputs
// ---------------------------------------------------------------------------
__global__ void __launch_bounds__(256) mix_cast_kernel(
    const float* __restrict__ hidden,
    const float* __restrict__ mk, const float* __restrict__ mv,
    const float* __restrict__ mr, const float* __restrict__ mg,
    __bf16* __restrict__ Xk, __bf16* __restrict__ Xv,
    __bf16* __restrict__ Xr, __bf16* __restrict__ Xg) {
  int idx = blockIdx.x * 256 + threadIdx.x;   // over SEQ*HIDN
  int s = idx >> 12;
  int c = idx & (HIDN - 1);
  float hc = hidden[idx];
  float hp = (s > 0) ? hidden[idx - HIDN] : 0.0f;
  float m;
  m = mk[c]; Xk[idx] = (__bf16)(hc * m + hp * (1.0f - m));
  m = mv[c]; Xv[idx] = (__bf16)(hc * m + hp * (1.0f - m));
  m = mr[c]; Xr[idx] = (__bf16)(hc * m + hp * (1.0f - m));
  m = mg[c]; Xg[idx] = (__bf16)(hc * m + hp * (1.0f - m));
}

// ---------------------------------------------------------------------------
// K2/K5: Y = X @ W^T   (X: MxK bf16 row-major, W: NxK bf16 row-major)
// Block = 8 waves; block tile M=64, N=128. Each wave owns one 16-col strip
// and 4 M-subtiles (4 WMMAs per k-step, B-frag reused across them).
// A tiles flow through 3 rotating LDS buffers fed by async global->LDS
// copies (one tile kept in flight across the barrier, s_wait_asynccnt 1);
// B fragments are software-pipelined in registers. mode 1 applies SiLU.
// ---------------------------------------------------------------------------
__global__ void __launch_bounds__(256) gemm_bf16_kernel(
    const __bf16* __restrict__ A, const __bf16* __restrict__ W,
    float* __restrict__ Out, int M, int N, int K, int mode) {
  __shared__ __attribute__((aligned(16))) __bf16 Atile[3][64][32];
  const int tid  = threadIdx.x;
  const int lane = tid & 31;
  const int wave = tid >> 5;
  const int m0 = blockIdx.y * 64;
  const int n0 = blockIdx.x * 128 + wave * 16;

  // per-thread async staging slot: one b128 of the 64x32 tile
  const int srow = tid >> 2;
  const int scol = (tid & 3) * 8;
  const __bf16* aRow = A + (size_t)(m0 + srow) * K + scol;

  // prime two tiles (k=0 -> buf0, k=32 -> buf1)
  async_copy_b128(aRow + 0,  &Atile[0][srow][scol]);
  async_copy_b128(aRow + 32, &Atile[1][srow][scol]);
  int ba = 0, bb = 1, bc = 2;

  // B fragment software pipeline
  const __bf16* pB =
      W + (size_t)(n0 + (lane & 15)) * K + ((lane >> 4) & 1) * 16;
  union { int4 q[2]; v16bf v; } bcur, bnext;
  bcur.q[0] = *(const int4*)(pB);
  bcur.q[1] = *(const int4*)(pB + 8);

  v8f zero = {0.f, 0.f, 0.f, 0.f, 0.f, 0.f, 0.f, 0.f};
  v8f acc[4] = {zero, zero, zero, zero};

  for (int k = 0; k < K; k += 32) {
    wait_async1();            // this k's tile resident; next may be in flight
    __syncthreads();

    // keep the async pipe full: stage tile k+64 into the retired buffer
    int kpf = k + 64;
    if (kpf > K - 32) kpf = K - 32;   // branchless clamp (redundant re-load)
    async_copy_b128(aRow + kpf, &Atile[bc][srow][scol]);

    // register-pipeline next B fragment
    const __bf16* pBn = pB + ((k + 32 < K) ? 32 : 0);
    bnext.q[0] = *(const int4*)(pBn);
    bnext.q[1] = *(const int4*)(pBn + 8);
    __builtin_prefetch((const void*)(pBn + 32), 0, 1);  // global_prefetch_b8

    // load all four A-frags, then issue the four WMMAs
    const __bf16* at = &Atile[ba][0][0];
    v16bf a0 = frag_a(at +  0 * 32, 32, lane);
    v16bf a1 = frag_a(at + 16 * 32, 32, lane);
    v16bf a2 = frag_a(at + 32 * 32, 32, lane);
    v16bf a3 = frag_a(at + 48 * 32, 32, lane);
    acc[0] = wmma_bf16(a0, bcur.v, acc[0]);
    acc[1] = wmma_bf16(a1, bcur.v, acc[1]);
    acc[2] = wmma_bf16(a2, bcur.v, acc[2]);
    acc[3] = wmma_bf16(a3, bcur.v, acc[3]);

    bcur = bnext;
    pB = pBn;
    int t = ba; ba = bb; bb = bc; bc = t;   // rotate LDS buffers
  }

  const int col  = lane & 15;
  const int rofs = (lane >> 4) * 8;
#pragma unroll
  for (int ms = 0; ms < 4; ++ms) {
#pragma unroll
    for (int r = 0; r < 8; ++r) {
      float v = acc[ms][r];
      if (mode == 1) v = v / (1.0f + __expf(-v));   // SiLU
      Out[(size_t)(m0 + ms * 16 + r + rofs) * N + n0 + col] = v;
    }
  }
}

// ---------------------------------------------------------------------------
// K3: chunked WKV scan. One block (4 waves) per head; wave w owns output
// columns e in [16w, 16w+16). Per-head state S (64x64 f32) lives entirely in
// WMMA accumulator registers (4 d-tiles of 16x16 per wave). Chunk r/k/v are
// pulled into double-buffered LDS by the Tensor Data Mover, pipelined one
// chunk ahead (s_wait_tensorcnt 3 keeps next chunk's 3 DMAs in flight).
// ---------------------------------------------------------------------------
__global__ void __launch_bounds__(128) rwkv_scan_kernel(
    const float* __restrict__ rbuf, const float* __restrict__ kbuf,
    const float* __restrict__ vbuf, const float* __restrict__ time_decay,
    const float* __restrict__ time_faaaa, float* __restrict__ att) {
  __shared__ __attribute__((aligned(16))) float  Rraw[2][CT][HD];
  __shared__ __attribute__((aligned(16))) float  Kraw[2][CT][HD];
  __shared__ __attribute__((aligned(16))) float  Vraw[2][CT][HD];
  __shared__ __attribute__((aligned(16))) __bf16 Rw[CT][HD];   // r_i * td^i
  __shared__ __attribute__((aligned(16))) __bf16 KwT[HD][CT];  // k_u*td^{31-u}
  __shared__ __attribute__((aligned(16))) __bf16 VbT[HD][CT];  // v  [e][u]
  __shared__ __attribute__((aligned(16))) __bf16 Mb[CT][CT];   // intra matrix
  __shared__ float  powL[CT + 1][HD];
  __shared__ float  tfL[HD];

  const int h    = blockIdx.x;
  const int tid  = threadIdx.x;
  const int lane = tid & 31;
  const int e0   = (tid >> 5) * 16;

  if (tid < HD) {
    float td = __expf(-__expf(time_decay[h * HD + tid]));
    float pw = 1.0f;
    for (int q = 0; q <= CT; ++q) { powL[q][tid] = pw; pw *= td; }
    tfL[tid] = time_faaaa[h * HD + tid];
  }

#if HAVE_TDM
  if (tid < 32) {   // wave 0 drives the Tensor Data Mover, chunk 0 -> buf 0
    const size_t g0ofs = (size_t)h * HD;
    tdm_load_2d_f32(rbuf + g0ofs, &Rraw[0][0][0], CT, HD, HIDN);
    tdm_load_2d_f32(kbuf + g0ofs, &Kraw[0][0][0], CT, HD, HIDN);
    tdm_load_2d_f32(vbuf + g0ofs, &Vraw[0][0][0], CT, HD, HIDN);
  }
#endif
  __syncthreads();

  v8f zero = {0.f, 0.f, 0.f, 0.f, 0.f, 0.f, 0.f, 0.f};
  v8f st[4] = {zero, zero, zero, zero};  // state columns [e0,e0+16)

  for (int c = 0; c < NCHUNK; ++c) {
    const int t0 = c * CT;
    const int pb = c & 1;

#if HAVE_TDM
    if (tid < 32) {
      if (c + 1 < NCHUNK) {   // issue next chunk, then wait for current
        const size_t gn = (size_t)(t0 + CT) * HIDN + (size_t)h * HD;
        tdm_load_2d_f32(rbuf + gn, &Rraw[pb ^ 1][0][0], CT, HD, HIDN);
        tdm_load_2d_f32(kbuf + gn, &Kraw[pb ^ 1][0][0], CT, HD, HIDN);
        tdm_load_2d_f32(vbuf + gn, &Vraw[pb ^ 1][0][0], CT, HD, HIDN);
        __builtin_amdgcn_s_wait_tensorcnt(3);
      } else {
        __builtin_amdgcn_s_wait_tensorcnt(0);
      }
    }
    __syncthreads();
#else
    for (int idx = tid; idx < CT * HD; idx += 128) {
      int i = idx >> 6, d = idx & 63;
      size_t g = (size_t)(t0 + i) * HIDN + (size_t)h * HD + d;
      Rraw[pb][i][d] = rbuf[g];
      Kraw[pb][i][d] = kbuf[g];
      Vraw[pb][i][d] = vbuf[g];
    }
    __syncthreads();
#endif

    // --- derive bf16 WMMA operands ---
    for (int idx = tid; idx < CT * HD; idx += 128) {
      int i = idx >> 6, d = idx & 63;
      Rw[i][d]  = (__bf16)(Rraw[pb][i][d] * powL[i][d]);
      KwT[d][i] = (__bf16)(Kraw[pb][i][d] * powL[CT - 1 - i][d]);
      VbT[d][i] = (__bf16)Vraw[pb][i][d];
    }
    __syncthreads();

    // --- intra-chunk 32x32 weighted Gram matrix M (VALU) ---
    for (int mi = 0; mi < 8; ++mi) {
      int e = mi * 128 + tid;
      int i = e >> 5, j = e & 31;
      float s = 0.0f;
      if (j == i) {
        for (int d = 0; d < HD; ++d)
          s += Rraw[pb][i][d] * tfL[d] * Kraw[pb][i][d];
      } else if (j < i) {
        const float* pw = &powL[i - 1 - j][0];
        for (int d = 0; d < HD; ++d)
          s += Rraw[pb][i][d] * pw[d] * Kraw[pb][j][d];
      }
      Mb[i][j] = (__bf16)s;
    }
    __syncthreads();

    // --- state -> B fragments (bf16, half-wave swap via ds_swizzle) ---
    v16bf s01 = state_bfrag(st[0], st[1], lane);   // K rows d = 0..31
    v16bf s23 = state_bfrag(st[2], st[3], lane);   // K rows d = 32..63
    v16bf vb  = frag_b(&VbT[e0][0], CT, lane);     // V chunk, this wave's e

    // --- O = Rw @ S0 + M @ V ---
    v8f o0 = zero, o1 = zero;                      // i-tiles 0..15 / 16..31
    o0 = wmma_bf16(frag_a(&Rw[0][0],   HD, lane), s01, o0);
    o1 = wmma_bf16(frag_a(&Rw[16][0],  HD, lane), s01, o1);
    o0 = wmma_bf16(frag_a(&Rw[0][32],  HD, lane), s23, o0);
    o1 = wmma_bf16(frag_a(&Rw[16][32], HD, lane), s23, o1);
    o0 = wmma_bf16(frag_a(&Mb[0][0],   CT, lane), vb, o0);
    o1 = wmma_bf16(frag_a(&Mb[16][0],  CT, lane), vb, o1);

    const int col  = lane & 15;
    const int rofs = (lane >> 4) * 8;
#pragma unroll
    for (int r = 0; r < 8; ++r) {
      size_t cbase = (size_t)h * HD + e0 + col;
      att[(size_t)(t0 + r + rofs) * HIDN + cbase]      = o0[r];
      att[(size_t)(t0 + 16 + r + rofs) * HIDN + cbase] = o1[r];
    }

    // --- state update: S = td^32 * S + KwT @ V ---
#pragma unroll
    for (int dt = 0; dt < 4; ++dt) {
      v16bf ka = frag_a(&KwT[dt * 16][0], CT, lane);
#pragma unroll
      for (int r = 0; r < 8; ++r)
        st[dt][r] *= powL[CT][dt * 16 + r + rofs];
      st[dt] = wmma_bf16(ka, vb, st[dt]);
    }
    __syncthreads();   // protect LDS before next chunk's staging
  }
}

// ---------------------------------------------------------------------------
// K4: per-(token, head) GroupNorm (groups of 64) + gate multiply -> bf16
// ---------------------------------------------------------------------------
__global__ void __launch_bounds__(256) gn_gate_kernel(
    const float* __restrict__ att, const float* __restrict__ gate,
    const float* __restrict__ gnw, const float* __restrict__ gnb,
    __bf16* __restrict__ Xo) {
  const int s    = blockIdx.x;
  const int wave = threadIdx.x >> 5;
  const int lane = threadIdx.x & 31;
  for (int hh = 0; hh < 8; ++hh) {
    const int hId  = wave * 8 + hh;
    const size_t b = (size_t)s * HIDN + (size_t)hId * HD;
    const int d0   = lane * 2;
    float x0 = att[b + d0], x1 = att[b + d0 + 1];
    float sm = x0 + x1;
    float sq = x0 * x0 + x1 * x1;
    for (int off = 16; off > 0; off >>= 1) {
      sm += __shfl_xor(sm, off, 32);
      sq += __shfl_xor(sq, off, 32);
    }
    float mu  = sm * (1.0f / 64.0f);
    float var = sq * (1.0f / 64.0f) - mu * mu;
    float inv = rsqrtf(var + 1e-5f);
    float y0 = (x0 - mu) * inv * gnw[hId * HD + d0]     + gnb[hId * HD + d0];
    float y1 = (x1 - mu) * inv * gnw[hId * HD + d0 + 1] + gnb[hId * HD + d0 + 1];
    Xo[b + d0]     = (__bf16)(y0 * gate[b + d0]);
    Xo[b + d0 + 1] = (__bf16)(y1 * gate[b + d0 + 1]);
  }
}

// ---------------------------------------------------------------------------
// Host-side launch
// ---------------------------------------------------------------------------
extern "C" void kernel_launch(void* const* d_in, const int* in_sizes, int n_in,
                              void* d_out, int out_size, void* d_ws,
                              size_t ws_size, hipStream_t stream) {
  const float* hidden = (const float*)d_in[0];
  const float* w_key  = (const float*)d_in[1];
  const float* w_val  = (const float*)d_in[2];
  const float* w_rec  = (const float*)d_in[3];
  const float* w_gate = (const float*)d_in[4];
  const float* w_out  = (const float*)d_in[5];
  const float* mk     = (const float*)d_in[6];
  const float* mv     = (const float*)d_in[7];
  const float* mr     = (const float*)d_in[8];
  const float* mg     = (const float*)d_in[9];
  const float* tdec   = (const float*)d_in[10];
  const float* tfaa   = (const float*)d_in[11];
  const float* gnw    = (const float*)d_in[12];
  const float* gnb    = (const float*)d_in[13];

  char* ws = (char*)d_ws;
  const size_t szW  = (size_t)HIDN * HIDN * 2;  // 32MB  bf16 weight
  const size_t szXb = (size_t)SEQ * HIDN * 2;   // 8MB   bf16 activation
  const size_t szF  = (size_t)SEQ * HIDN * 4;   // 16MB  f32 activation

  __bf16* Wk = (__bf16*)(ws + 0 * szW);
  __bf16* Wv = (__bf16*)(ws + 1 * szW);
  __bf16* Wr = (__bf16*)(ws + 2 * szW);
  __bf16* Wg = (__bf16*)(ws + 3 * szW);
  __bf16* Wo = (__bf16*)(ws + 4 * szW);
  char* xbase = ws + 5 * szW;
  __bf16* Xk = (__bf16*)(xbase + 0 * szXb);
  __bf16* Xv = (__bf16*)(xbase + 1 * szXb);
  __bf16* Xr = (__bf16*)(xbase + 2 * szXb);
  __bf16* Xg = (__bf16*)(xbase + 3 * szXb);
  char* fbase = xbase + 4 * szXb;
  float* kb = (float*)(fbase + 0 * szF);
  float* vb = (float*)(fbase + 1 * szF);
  float* rb = (float*)(fbase + 2 * szF);
  float* gb = (float*)(fbase + 3 * szF);
  // aliases over regions that are dead by the time these are written:
  float*  att = (float*)(xbase);             // over Xk+Xv (16MB)
  __bf16* Xo  = (__bf16*)(xbase + 2 * szXb); // over Xr   (8MB)

  // K0: weights -> bf16 (5x)
  const int castGrid = (HIDN * HIDN) / 1024;
  cast_bf16_kernel<<<castGrid, 256, 0, stream>>>(w_key,  Wk);
  cast_bf16_kernel<<<castGrid, 256, 0, stream>>>(w_val,  Wv);
  cast_bf16_kernel<<<castGrid, 256, 0, stream>>>(w_rec,  Wr);
  cast_bf16_kernel<<<castGrid, 256, 0, stream>>>(w_gate, Wg);
  cast_bf16_kernel<<<castGrid, 256, 0, stream>>>(w_out,  Wo);

  // K1: token-shift mixing
  mix_cast_kernel<<<(SEQ * HIDN) / 256, 256, 0, stream>>>(
      hidden, mk, mv, mr, mg, Xk, Xv, Xr, Xg);

  // K2: four projection GEMMs (SiLU fused for gate)
  dim3 gg(HIDN / 128, SEQ / 64);
  gemm_bf16_kernel<<<gg, 256, 0, stream>>>(Xk, Wk, kb, SEQ, HIDN, HIDN, 0);
  gemm_bf16_kernel<<<gg, 256, 0, stream>>>(Xv, Wv, vb, SEQ, HIDN, HIDN, 0);
  gemm_bf16_kernel<<<gg, 256, 0, stream>>>(Xr, Wr, rb, SEQ, HIDN, HIDN, 0);
  gemm_bf16_kernel<<<gg, 256, 0, stream>>>(Xg, Wg, gb, SEQ, HIDN, HIDN, 1);

  // K3: chunked WKV scan, one block per head
  rwkv_scan_kernel<<<NH, 128, 0, stream>>>(rb, kb, vb, tdec, tfaa, att);

  // K4: GroupNorm + gate
  gn_gate_kernel<<<SEQ, 256, 0, stream>>>(att, gb, gnw, gnb, Xo);

  // K5: output projection into d_out (f32)
  gemm_bf16_kernel<<<gg, 256, 0, stream>>>(Xo, Wo, (float*)d_out, SEQ, HIDN,
                                           HIDN, 0);
  (void)in_sizes; (void)n_in; (void)out_size; (void)ws_size;
}
